// FP4Linear_17016660427503
// MI455X (gfx1250) — compile-verified, matrix operand order checked by reading, not run on hardware
//
#include <hip/hip_runtime.h>
#include <hip/hip_bf16.h>

// ---------------------------------------------------------------------------
// FP4Linear on MI455X (gfx1250):
//   y = x @ fp4_roundtrip(w)^T + bias
//   M=8192 K=4096 N=11008  -> 739 GFLOP, ~675MB traffic -> compute bound,
//   so the work runs on the WMMA pipes (v_wmma_f32_16x16x32_f16) and the
//   tile feed uses the CDNA5 async global->LDS DMA path (ASYNCcnt), which
//   removes the register-staging buffer that previously spilled to scratch.
// ---------------------------------------------------------------------------

typedef _Float16 half2t __attribute__((ext_vector_type(2)));
typedef _Float16 half4t __attribute__((ext_vector_type(4)));
typedef _Float16 half8  __attribute__((ext_vector_type(8)));
typedef _Float16 v16h   __attribute__((ext_vector_type(16)));
typedef float    v8f    __attribute__((ext_vector_type(8)));

// ------------------------- FP4 round-trip pre-pass -------------------------
// bnb FP4: 16 signed codes, blockwise absmax over 64 consecutive elements.
// searchsorted(bounds, n) with midpoint bounds == nearest-code (ties down),
// implemented as a monotone select chain (15 v_cndmask, stays in VALU regs).
__device__ __forceinline__ float fp4_nearest_code(float n) {
    const float c1  = -0.66666667f, c2  = -0.5f,        c3  = -0.33333333f;
    const float c4  = -0.25f,       c5  = -0.16666667f, c6  = -0.0052083333f;
    const float c7  = -0.0f,        c8  =  0.0f,        c9  =  0.0052083333f;
    const float c10 =  0.16666667f, c11 =  0.25f,       c12 =  0.33333333f;
    const float c13 =  0.5f,        c14 =  0.66666667f, c15 =  1.0f;
    float code = -1.0f;
    code = (n > -0.83333333f)   ? c1  : code;
    code = (n > -0.58333333f)   ? c2  : code;
    code = (n > -0.41666667f)   ? c3  : code;
    code = (n > -0.29166667f)   ? c4  : code;
    code = (n > -0.20833333f)   ? c5  : code;
    code = (n > -0.0859375f)    ? c6  : code;
    code = (n > -0.0026041667f) ? c7  : code;
    code = (n >  0.0f)          ? c8  : code;
    code = (n >  0.0f)          ? c9  : code;
    code = (n >  0.0026041667f) ? c9  : code;
    code = (n >  0.0859375f)    ? c10 : code;
    code = (n >  0.20833333f)   ? c11 : code;
    code = (n >  0.29166667f)   ? c12 : code;
    code = (n >  0.41666667f)   ? c13 : code;
    code = (n >  0.58333333f)   ? c14 : code;
    code = (n >  0.83333333f)   ? c15 : code;
    return code;
}

// One wave32 per 64-element block: lane holds 2 consecutive floats,
// absmax via 5-step __shfl_xor butterfly (wave32), dequant -> f16.
__global__ void fp4_quant_dequant_f16(const float* __restrict__ w,
                                      _Float16* __restrict__ wh,
                                      long long nblocks) {
    long long wave = (long long)blockIdx.x * (blockDim.x / 32) + (threadIdx.x >> 5);
    if (wave >= nblocks) return;
    int lane = threadIdx.x & 31;
    long long base = wave * 64 + (long long)lane * 2;

    float2 v = *(const float2*)(w + base);
    float am = fmaxf(fabsf(v.x), fabsf(v.y));
    #pragma unroll
    for (int off = 16; off > 0; off >>= 1)
        am = fmaxf(am, __shfl_xor(am, off, 32));
    float scale = fmaxf(am, 1e-12f);

    float d0 = fp4_nearest_code(v.x / scale) * scale;
    float d1 = fp4_nearest_code(v.y / scale) * scale;
    half2t h; h.x = (_Float16)d0; h.y = (_Float16)d1;
    *(half2t*)(wh + base) = h;
}

// ----------------------------- x : f32 -> f16 ------------------------------
__global__ void f32_to_f16_vec4(const float* __restrict__ x,
                                _Float16* __restrict__ xh,
                                long long n) {
    long long i = ((long long)blockIdx.x * blockDim.x + threadIdx.x) * 4;
    long long stride = (long long)gridDim.x * blockDim.x * 4;
    for (; i < n; i += stride) {
        float4 v = *(const float4*)(x + i);
        half4t h;
        h.x = (_Float16)v.x; h.y = (_Float16)v.y;
        h.z = (_Float16)v.z; h.w = (_Float16)v.w;
        *(half4t*)(xh + i) = h;
    }
}

// ------------------------------- WMMA GEMM ---------------------------------
// Workgroup tile 128(M) x 128(N), BK=64, 256 threads = 8 waves (4 M x 2 N),
// wave tile 32x64 -> 2x4 accumulator frags -> 16 v_wmma per wave per K-step.
// LDS double buffered: 2 * (A 128x72 + B 128x72) halves = 72 KB.
// Tiles are fed with global_load_async_to_lds_b128 (ASYNCcnt) so no staging
// VGPRs are live across the WMMA block.
#define BM 128
#define BN 128
#define BK 64
#define LDS_STRIDE 72                    // 64 + 8 halves pad (144B rows)
#define TILE_HALVES (BM * LDS_STRIDE)    // 9216 halves per tile
#define TILE_BYTES  (TILE_HALVES * 2)    // 18432 B per tile

__device__ __forceinline__ v16h load_frag(const _Float16* tile,
                                          int row_base, int kb, int lane) {
    int part = lane >> 4;        // which K half-group this lane owns
    int r    = lane & 15;        // matrix row (A: M, B: N) within the frag
    const _Float16* p = tile + (row_base + r) * LDS_STRIDE + kb + part * 8;
    half8 lo = *(const half8*)(p);        // K = 0..7   (or 8..15)
    half8 hi = *(const half8*)(p + 16);   // K = 16..23 (or 24..31)
    return __builtin_shufflevector(lo, hi, 0,1,2,3,4,5,6,7,8,9,10,11,12,13,14,15);
}

// 64B per thread: 4 async b128 copies. The 24-bit inst offset is added to
// BOTH the LDS and the global address (ISA 08_async_tensor 4.4), so
// offset:16/32/48 advances both sides of the copy.
//
// IMPORTANT: smem's generic pointer is passed as a dummy operand so the
// shared array ESCAPES into the asm. Without it, alias analysis proves the
// asm cannot write LDS, the fragment ds_loads read "uninitialized" memory,
// and the whole compute folds to poison (observed: all WMMA reading v[0:7]).
__device__ __forceinline__ void async_copy64(unsigned lds_off, const void* gaddr,
                                             void* lds_base) {
    asm volatile(
        "global_load_async_to_lds_b128 %0, %1, off\n\t"
        "global_load_async_to_lds_b128 %0, %1, off offset:16\n\t"
        "global_load_async_to_lds_b128 %0, %1, off offset:32\n\t"
        "global_load_async_to_lds_b128 %0, %1, off offset:48"
        :: "v"(lds_off), "v"(gaddr), "v"(lds_base) : "memory");
}

// Copy one 128x64-half A tile and one 128x64-half B tile into LDS buffer
// `buf` (0/1). Thread t owns row r = t/2, 64B half hs = t%2 of the 128B row.
__device__ __forceinline__ void async_copy_tiles(const _Float16* __restrict__ xh,
                                                 const _Float16* __restrict__ wh,
                                                 _Float16* smem,
                                                 int buf, int m0, int n0, int k0,
                                                 int K, int tid) {
    int r  = tid >> 1;
    int hs = tid & 1;
    unsigned lrow = (unsigned)(r * LDS_STRIDE + hs * 32) * 2u;   // bytes
    unsigned la = (unsigned)buf * 2u * TILE_BYTES + lrow;        // A tile
    unsigned lb = la + TILE_BYTES;                               // B tile
    const void* ga = (const void*)(xh + (long long)(m0 + r) * K + k0 + hs * 32);
    const void* gb = (const void*)(wh + (long long)(n0 + r) * K + k0 + hs * 32);
    async_copy64(la, ga, (void*)smem);
    async_copy64(lb, gb, (void*)smem);
}

__device__ __forceinline__ void wait_async0() {
    asm volatile("s_wait_asynccnt 0x0" ::: "memory");
}

__global__ __launch_bounds__(256)
void fp4linear_wmma_gemm(const _Float16* __restrict__ xh,   // [M,K] f16
                         const _Float16* __restrict__ wh,   // [N,K] f16 (B^T rows)
                         const float* __restrict__ bias,    // [N]
                         float* __restrict__ out,           // [M,N]
                         int M, int N, int K) {
    // Single shared array -> sits at workgroup LDS base (byte offset 0),
    // which the async-copy LDS offsets rely on.
    __shared__ _Float16 smem[4 * TILE_HALVES];  // [buf0 A,B][buf1 A,B] 73728 B

    int tid  = threadIdx.x;
    int lane = tid & 31;
    int wave = tid >> 5;
    int wm   = wave & 3;          // 0..3 -> M offset wm*32
    int wn   = wave >> 2;         // 0..1 -> N offset wn*64

    int m0 = blockIdx.y * BM;
    int n0 = blockIdx.x * BN;

    v8f acc[2][4];
    #pragma unroll
    for (int i = 0; i < 2; ++i)
        #pragma unroll
        for (int j = 0; j < 4; ++j)
            acc[i][j] = (v8f)(0.0f);

    int nk = K / BK;

    async_copy_tiles(xh, wh, smem, /*buf=*/0, m0, n0, 0, K, tid);
    wait_async0();
    __syncthreads();

    for (int kt = 0; kt < nk; ++kt) {
        int cur = kt & 1;
        if (kt + 1 < nk)
            async_copy_tiles(xh, wh, smem, 1 - cur, m0, n0, (kt + 1) * BK, K, tid);

        const _Float16* As = smem + cur * 2 * TILE_HALVES;
        const _Float16* Bs = As + TILE_HALVES;

        #pragma unroll
        for (int ks = 0; ks < 2; ++ks) {       // two 16x16x32 K-slices of BK=64
            v16h a0 = load_frag(As, wm * 32,      ks * 32, lane);
            v16h a1 = load_frag(As, wm * 32 + 16, ks * 32, lane);
            #pragma unroll
            for (int j = 0; j < 4; ++j) {
                v16h b = load_frag(Bs, wn * 64 + j * 16, ks * 32, lane);
                acc[0][j] = __builtin_amdgcn_wmma_f32_16x16x32_f16(
                    false, a0, false, b, (short)0, acc[0][j], false, false);
                acc[1][j] = __builtin_amdgcn_wmma_f32_16x16x32_f16(
                    false, a1, false, b, (short)0, acc[1][j], false, false);
            }
        }

        // Own-wave async copies into the next buffer must be complete before
        // the barrier publishes it to the other waves.
        wait_async0();
        __syncthreads();
    }

    // Epilogue per ISA 7.12.2 f32 C/D layout:
    //   VGPR v, lanes 0-15: M=v, N=lane ; lanes 16-31: M=v+8, N=lane-16.
    int part  = lane >> 4;
    int col16 = lane & 15;
    #pragma unroll
    for (int i = 0; i < 2; ++i) {
        #pragma unroll
        for (int j = 0; j < 4; ++j) {
            int gm0 = m0 + wm * 32 + i * 16 + part * 8;
            int gn  = n0 + wn * 64 + j * 16 + col16;
            float bv = bias[gn];
            #pragma unroll
            for (int v = 0; v < 8; ++v)
                out[(long long)(gm0 + v) * N + gn] = acc[i][j][v] + bv;
        }
    }
}

// ------------------------------- launcher ----------------------------------
extern "C" void kernel_launch(void* const* d_in, const int* in_sizes, int n_in,
                              void* d_out, int out_size, void* d_ws, size_t ws_size,
                              hipStream_t stream) {
    const float* x    = (const float*)d_in[0];   // [M,K] f32
    const float* w    = (const float*)d_in[1];   // [N,K] f32
    const float* bias = (const float*)d_in[2];   // [N]  f32
    float* out = (float*)d_out;

    long long xn = (long long)in_sizes[0];       // M*K
    long long wn = (long long)in_sizes[1];       // N*K
    int N = in_sizes[2];
    int K = (int)(wn / N);
    int M = (int)(xn / K);

    _Float16* wh = (_Float16*)d_ws;              // N*K f16
    _Float16* xh = wh + wn;                      // M*K f16  (total ~157 MB)

    // 1) FP4 round-trip weights -> f16 (one wave32 per 64-elem block)
    long long nblocks = wn / 64;
    int qgrid = (int)((nblocks + 7) / 8);        // 8 waves per 256-thread WG
    fp4_quant_dequant_f16<<<qgrid, 256, 0, stream>>>(w, wh, nblocks);

    // 2) x -> f16
    f32_to_f16_vec4<<<4096, 256, 0, stream>>>(x, xh, xn);

    // 3) WMMA GEMM (shapes divide tiles exactly: 8192/128, 11008/128, 4096/64)
    dim3 grid(N / BN, M / BM);
    fp4linear_wmma_gemm<<<grid, 256, 0, stream>>>(xh, wh, bias, out, M, N, K);
}
